// GroupedQueryAttention_21423296873306
// MI455X (gfx1250) — matrix operand order repaired and checked by reading, not compiled, not run on hardware
//
#include <hip/hip_runtime.h>
#include <stdint.h>

typedef __bf16 bf16;
typedef __attribute__((ext_vector_type(16))) __bf16 v16bf;
typedef __attribute__((ext_vector_type(8)))  float  v8f;

#define B_  2
#define S_  2048
#define D_  2048
#define H_  32
#define G_  8
#define HD_ 64
#define GS_ (H_ / G_)

union FragU { v16bf v; unsigned u[8]; };

__device__ inline v8f v8f_zero() {
  v8f z;
#pragma unroll
  for (int i = 0; i < 8; ++i) z[i] = 0.0f;
  return z;
}

// ---- CDNA5 async global->LDS copy (ASYNCcnt-tracked, no VGPR data path) ----
__device__ inline void async_g2l_b128(unsigned lds_off, const bf16* g) {
  asm volatile("global_load_async_to_lds_b128 %0, %1, off"
               :: "v"(lds_off), "v"((unsigned long long)(uintptr_t)g)
               : "memory");
}
__device__ inline void wait_async0() {
#if __has_builtin(__builtin_amdgcn_s_wait_asynccnt)
  __builtin_amdgcn_s_wait_asynccnt(0);
#else
  asm volatile("s_wait_asynccnt 0x0" ::: "memory");
#endif
}

// ---------------------------------------------------------------- converts
__global__ void k_f32_to_bf16(const float* __restrict__ in, bf16* __restrict__ out, long n) {
  long i = (long)blockIdx.x * blockDim.x + threadIdx.x;
  long st = (long)gridDim.x * blockDim.x;
  for (; i < n; i += st) out[i] = (bf16)in[i];
}

// W[k][n] (f32, KxN) -> Wt[n][k] (bf16, NxK)
__global__ void k_transpose_bf16(const float* __restrict__ W, bf16* __restrict__ Wt, int K, int N) {
  long total = (long)K * N;
  long st = (long)gridDim.x * blockDim.x;
  for (long i = (long)blockIdx.x * blockDim.x + threadIdx.x; i < total; i += st) {
    long k = i / N, n = i % N;
    Wt[n * (long)K + k] = (bf16)W[i];
  }
}

// v_raw [(b*S+s)*G*HD + g*HD + d] f32 -> Vt [((b*G+g)*HD + d)*S + s] bf16
__global__ void k_v_pack(const float* __restrict__ vraw, bf16* __restrict__ vt) {
  long total = (long)B_ * S_ * G_ * HD_;
  long st = (long)gridDim.x * blockDim.x;
  for (long i = (long)blockIdx.x * blockDim.x + threadIdx.x; i < total; i += st) {
    int d = (int)(i & 63);
    int g = (int)((i >> 6) & 7);
    int s = (int)((i >> 9) & 2047);
    int b = (int)(i >> 20);
    vt[(((long)b * G_ + g) * HD_ + d) * S_ + s] = (bf16)vraw[i];
  }
}

// ---------------------------------------------------------------- GEMM
// C[M,N] f32 = A[M,K] bf16 (row major) x Bt[N,K] bf16 (B transposed).
// Block 256 thr = 8 waves; block tile 128x128; wave tile 32x64; K-step 32.
// LDS double-buffered; panels staged with global_load_async_to_lds_b128 so the
// copy of panel t+1 overlaps WMMA compute on panel t.
#define GBM 128
#define GBN 128
#define GBK 32
#define LDSP 48  // inner LDS stride (bf16), 96B rows keep 16B alignment

__global__ __launch_bounds__(256)
void k_gemm_bf16(const bf16* __restrict__ A, const bf16* __restrict__ Bt,
                 float* __restrict__ C, int M, int N, int K) {
  __shared__ bf16 As[2][GBM][LDSP];
  __shared__ bf16 Bs[2][GBN][LDSP];
  const int tid  = threadIdx.x;
  const int lane = tid & 31;
  const int wave = tid >> 5;
  const int wm = wave >> 1, wn = wave & 1;
  const int bm = blockIdx.y * GBM;
  const int bn = blockIdx.x * GBN;
  const int nlo = lane & 15;
  const int kh  = lane >> 4;

  v8f acc[2][4];
#pragma unroll
  for (int mt = 0; mt < 2; ++mt)
#pragma unroll
    for (int nt = 0; nt < 4; ++nt) acc[mt][nt] = v8f_zero();

  // per-thread chunk coordinates (512 chunks of 8 bf16 per panel)
  auto stage = [&](int buf, int kb) {
#pragma unroll
    for (int t = 0; t < 2; ++t) {
      int c = tid + t * 256;
      int row = c >> 2;
      int col = (c & 3) * 8;
      async_g2l_b128((unsigned)(unsigned long long)(uintptr_t)&As[buf][row][col],
                     A + (long)(bm + row) * K + kb + col);
      async_g2l_b128((unsigned)(unsigned long long)(uintptr_t)&Bs[buf][row][col],
                     Bt + (long)(bn + row) * K + kb + col);
    }
  };

  stage(0, 0);
  wait_async0();
  __syncthreads();

  int cur = 0;
  for (int kb = 0; kb < K; kb += GBK) {
    if (kb + GBK < K) stage(cur ^ 1, kb + GBK);   // overlap copy with compute

    FragU a[2], b[4];
#pragma unroll
    for (int mt = 0; mt < 2; ++mt) {
      int m = wm * 32 + mt * 16 + nlo;
#pragma unroll
      for (int i = 0; i < 8; ++i) {
        int kk = ((i >> 2) << 4) + (kh << 3) + ((i & 3) << 1);
        a[mt].u[i] = *(const unsigned*)&As[cur][m][kk];
      }
    }
#pragma unroll
    for (int nt = 0; nt < 4; ++nt) {
      int nn = wn * 64 + nt * 16 + nlo;
#pragma unroll
      for (int i = 0; i < 8; ++i) {
        int kk = ((i >> 2) << 4) + (kh << 3) + ((i & 3) << 1);
        b[nt].u[i] = *(const unsigned*)&Bs[cur][nn][kk];
      }
    }
#pragma unroll
    for (int mt = 0; mt < 2; ++mt)
#pragma unroll
      for (int nt = 0; nt < 4; ++nt)
        acc[mt][nt] = __builtin_amdgcn_wmma_f32_16x16x32_bf16(
            false, a[mt].v, false, b[nt].v, (short)0, acc[mt][nt], false, false);

    wait_async0();      // my async stores into buf^1 have landed in LDS
    __syncthreads();    // everyone's have; everyone done reading buf
    cur ^= 1;
  }

#pragma unroll
  for (int mt = 0; mt < 2; ++mt)
#pragma unroll
    for (int nt = 0; nt < 4; ++nt)
#pragma unroll
      for (int r = 0; r < 8; ++r) {
        int row = bm + wm * 32 + mt * 16 + r + (kh << 3);
        int col = bn + wn * 64 + nt * 16 + nlo;
        C[(long)row * N + col] = acc[mt][nt][r];
      }
}

// ---------------------------------------------------------------- RMSNorm + RoPE
// raw: [(b*S+s)*nheads*64 + hh*64 + d] f32 -> out: [((b*nheads+hh)*S+s)*64 + d] bf16
__global__ __launch_bounds__(256)
void k_norm_rope(const float* __restrict__ raw, const float* __restrict__ cost,
                 const float* __restrict__ sint, const float* __restrict__ w,
                 bf16* __restrict__ out, int nheads) {
  int wave = threadIdx.x >> 5;
  int lane = threadIdx.x & 31;
  long rowid = (long)blockIdx.x * 8 + wave;
  long nrows = (long)B_ * S_ * nheads;
  if (rowid >= nrows) return;
  const float* xr = raw + rowid * HD_;
  float x1 = xr[lane], x2 = xr[lane + 32];
  float ss = x1 * x1 + x2 * x2;
#pragma unroll
  for (int m = 16; m >= 1; m >>= 1) ss += __shfl_xor(ss, m, 32);
  float inv = rsqrtf(ss * (1.0f / HD_) + 1e-6f);
  float y1 = x1 * inv * w[lane];
  float y2 = x2 * inv * w[lane + 32];
  int b  = (int)(rowid / ((long)S_ * nheads));
  int rm = (int)(rowid % ((long)S_ * nheads));
  int s  = rm / nheads;
  int hh = rm % nheads;
  const float* cp = cost + (long)s * HD_;
  const float* sp = sint + (long)s * HD_;
  float o1 = y1 * cp[lane]      - y2 * sp[lane];        // rot = [-x2, x1]
  float o2 = y2 * cp[lane + 32] + y1 * sp[lane + 32];
  bf16* op = out + (((long)b * nheads + hh) * S_ + s) * HD_;
  op[lane]      = (bf16)o1;
  op[lane + 32] = (bf16)o2;
}

// ---------------------------------------------------------------- flash attention
// grid (S/64, B*H), 128 thr = 4 waves; each wave: 16 queries x HD=64 output.
__global__ __launch_bounds__(128)
void k_attn(const bf16* __restrict__ Q, const bf16* __restrict__ Kc,
            const bf16* __restrict__ Vt, bf16* __restrict__ ctx) {
  __shared__ bf16 P[4][16][36];  // per-wave P relayout slab (rows 72B, dword aligned)
  const int lane = threadIdx.x & 31;
  const int wave = threadIdx.x >> 5;
  const int nlo = lane & 15;
  const int kh  = lane >> 4;

  const int bh = blockIdx.y;
  const int b = bh / H_, h = bh % H_;
  const int g = h / GS_;
  const int qbase = (blockIdx.x * 4 + wave) * 16;

  const bf16* Qp = Q  + ((long)b * H_ + h) * S_ * HD_;
  const bf16* Kp = Kc + ((long)b * G_ + g) * S_ * HD_;
  const bf16* Vp = Vt + ((long)b * G_ + g) * HD_ * S_;

  FragU a0, a1;  // q fragment, K-dims 0..31 and 32..63 (reused all tiles)
  {
    int m = qbase + nlo;
#pragma unroll
    for (int i = 0; i < 8; ++i) {
      int kk = ((i >> 2) << 4) + (kh << 3) + ((i & 3) << 1);
      a0.u[i] = *(const unsigned*)(Qp + (long)m * HD_ + kk);
      a1.u[i] = *(const unsigned*)(Qp + (long)m * HD_ + 32 + kk);
    }
  }

  v8f o[4];
#pragma unroll
  for (int i = 0; i < 4; ++i) o[i] = v8f_zero();
  float rowm[8], rowl[8];
#pragma unroll
  for (int r = 0; r < 8; ++r) { rowm[r] = -1e30f; rowl[r] = 0.0f; }

  const int kend = qbase + 16;
  for (int kb = 0; kb < kend; kb += 32) {
    // ---- scores S = (Q Kt)/8 for 16q x 32k
    v8f s0 = v8f_zero(), s1 = v8f_zero();
    FragU bk;
#pragma unroll
    for (int nt = 0; nt < 2; ++nt) {
      int key = kb + nt * 16 + nlo;
#pragma unroll
      for (int ks = 0; ks < 2; ++ks) {
#pragma unroll
        for (int i = 0; i < 8; ++i)
          bk.u[i] = *(const unsigned*)(Kp + (long)key * HD_ + ks * 32 + ((i >> 2) << 4) + (kh << 3) + ((i & 3) << 1));
        if (nt == 0)
          s0 = __builtin_amdgcn_wmma_f32_16x16x32_bf16(false, ks ? a1.v : a0.v,
                                                       false, bk.v, (short)0, s0, false, false);
        else
          s1 = __builtin_amdgcn_wmma_f32_16x16x32_bf16(false, ks ? a1.v : a0.v,
                                                       false, bk.v, (short)0, s1, false, false);
      }
    }
    // ---- causal mask + online softmax (row m = r + 8*kh, col = lane&15)
#pragma unroll
    for (int r = 0; r < 8; ++r) {
      int qi = qbase + r + (kh << 3);
      float v0 = s0[r] * 0.125f; if (kb + nlo > qi)      v0 = -1e30f;
      float v1 = s1[r] * 0.125f; if (kb + 16 + nlo > qi) v1 = -1e30f;
      float mx = fmaxf(v0, v1);
      mx = fmaxf(mx, __shfl_xor(mx, 1, 32));
      mx = fmaxf(mx, __shfl_xor(mx, 2, 32));
      mx = fmaxf(mx, __shfl_xor(mx, 4, 32));
      mx = fmaxf(mx, __shfl_xor(mx, 8, 32));
      float mnew  = fmaxf(rowm[r], mx);
      float alpha = __expf(rowm[r] - mnew);
      rowm[r] = mnew;
      float p0 = __expf(v0 - mnew);
      float p1 = __expf(v1 - mnew);
      s0[r] = p0; s1[r] = p1;
      float rs = p0 + p1;
      rs += __shfl_xor(rs, 1, 32);
      rs += __shfl_xor(rs, 2, 32);
      rs += __shfl_xor(rs, 4, 32);
      rs += __shfl_xor(rs, 8, 32);
      rowl[r] = rowl[r] * alpha + rs;
#pragma unroll
      for (int nt2 = 0; nt2 < 4; ++nt2) o[nt2][r] *= alpha;
    }
    // ---- P: C-frag -> LDS -> A-frag (wave-private slab)
#pragma unroll
    for (int r = 0; r < 8; ++r) {
      P[wave][r + (kh << 3)][nlo]      = (bf16)s0[r];
      P[wave][r + (kh << 3)][16 + nlo] = (bf16)s1[r];
    }
    FragU pf;
#pragma unroll
    for (int i = 0; i < 8; ++i) {
      int kk = ((i >> 2) << 4) + (kh << 3) + ((i & 3) << 1);
      pf.u[i] = *(const unsigned*)&P[wave][nlo][kk];
    }
    // ---- ctx += P @ V  (Vt is [hd][s], so B-frag loads are K-contiguous)
    FragU vf;
#pragma unroll
    for (int nt2 = 0; nt2 < 4; ++nt2) {
      int hd = nt2 * 16 + nlo;
#pragma unroll
      for (int i = 0; i < 8; ++i)
        vf.u[i] = *(const unsigned*)(Vp + (long)hd * S_ + kb + ((i >> 2) << 4) + (kh << 3) + ((i & 3) << 1));
      o[nt2] = __builtin_amdgcn_wmma_f32_16x16x32_bf16(false, pf.v, false, vf.v,
                                                       (short)0, o[nt2], false, false);
    }
  }
  // ---- normalize and store ctx as [b*S+s][h*64+hd] bf16
#pragma unroll
  for (int nt2 = 0; nt2 < 4; ++nt2)
#pragma unroll
    for (int r = 0; r < 8; ++r) {
      int srow = qbase + r + (kh << 3);
      int col  = h * HD_ + nt2 * 16 + nlo;
      ctx[((long)b * S_ + srow) * (H_ * HD_) + col] = (bf16)(o[nt2][r] / rowl[r]);
    }
}

// ---------------------------------------------------------------- launch
extern "C" void kernel_launch(void* const* d_in, const int* in_sizes, int n_in,
                              void* d_out, int out_size, void* d_ws, size_t ws_size,
                              hipStream_t stream) {
  const float* x    = (const float*)d_in[0];
  // d_in[1] = mask (causality implemented directly)
  const float* cost = (const float*)d_in[2];
  const float* sint = (const float*)d_in[3];
  const float* Wq   = (const float*)d_in[4];
  const float* Wk   = (const float*)d_in[5];
  const float* Wv   = (const float*)d_in[6];
  const float* Wo   = (const float*)d_in[7];
  const float* qw   = (const float*)d_in[8];
  const float* kw   = (const float*)d_in[9];
  (void)in_sizes; (void)n_in; (void)out_size; (void)ws_size;

  const long MR = (long)B_ * S_;          // 4096
  char* p = (char*)d_ws;
  auto carve = [&](size_t bytes) { char* r = p; p += (bytes + 255) & ~(size_t)255; return r; };

  bf16* xb   = (bf16*)carve(MR * D_ * sizeof(bf16));
  bf16* WqT  = (bf16*)carve((size_t)(H_ * HD_) * D_ * sizeof(bf16));
  bf16* WkT  = (bf16*)carve((size_t)(G_ * HD_) * D_ * sizeof(bf16));
  bf16* WvT  = (bf16*)carve((size_t)(G_ * HD_) * D_ * sizeof(bf16));
  bf16* WoT  = (bf16*)carve((size_t)D_ * (H_ * HD_) * sizeof(bf16));
  float* qraw = (float*)carve(MR * (H_ * HD_) * sizeof(float));
  float* kraw = (float*)carve(MR * (G_ * HD_) * sizeof(float));
  float* vraw = (float*)carve(MR * (G_ * HD_) * sizeof(float));
  bf16* Qb   = (bf16*)carve((size_t)B_ * H_ * S_ * HD_ * sizeof(bf16));
  bf16* Kb   = (bf16*)carve((size_t)B_ * G_ * S_ * HD_ * sizeof(bf16));
  bf16* Vtb  = (bf16*)carve((size_t)B_ * G_ * HD_ * S_ * sizeof(bf16));
  bf16* ctxb = (bf16*)carve(MR * (H_ * HD_) * sizeof(bf16));

  k_f32_to_bf16<<<512, 256, 0, stream>>>(x, xb, MR * D_);
  k_transpose_bf16<<<512, 256, 0, stream>>>(Wq, WqT, D_, H_ * HD_);
  k_transpose_bf16<<<256, 256, 0, stream>>>(Wk, WkT, D_, G_ * HD_);
  k_transpose_bf16<<<256, 256, 0, stream>>>(Wv, WvT, D_, G_ * HD_);
  k_transpose_bf16<<<512, 256, 0, stream>>>(Wo, WoT, H_ * HD_, D_);

  dim3 gq((H_ * HD_) / GBN, MR / GBM);       // (16, 32)
  dim3 gk((G_ * HD_) / GBN, MR / GBM);       // (4, 32)
  k_gemm_bf16<<<gq, 256, 0, stream>>>(xb, WqT, qraw, (int)MR, H_ * HD_, D_);
  k_gemm_bf16<<<gk, 256, 0, stream>>>(xb, WkT, kraw, (int)MR, G_ * HD_, D_);
  k_gemm_bf16<<<gk, 256, 0, stream>>>(xb, WvT, vraw, (int)MR, G_ * HD_, D_);

  k_norm_rope<<<(B_ * S_ * H_) / 8, 256, 0, stream>>>(qraw, cost, sint, qw, Qb, H_);
  k_norm_rope<<<(B_ * S_ * G_) / 8, 256, 0, stream>>>(kraw, cost, sint, kw, Kb, G_);
  k_v_pack<<<512, 256, 0, stream>>>(vraw, Vtb);

  dim3 ga(S_ / 64, B_ * H_);                  // (32, 64)
  k_attn<<<ga, 128, 0, stream>>>(Qb, Kb, Vtb, ctxb);

  k_gemm_bf16<<<gq, 256, 0, stream>>>(ctxb, WoT, (float*)d_out, (int)MR, D_, H_ * HD_);
}
